// FastNGPNeRF_47974784697096
// MI455X (gfx1250) — matrix operand order repaired and checked by reading, not compiled, not run on hardware
//
#include <hip/hip_runtime.h>
#include <hip/hip_bf16.h>
#include <math.h>

// ---------------------------------------------------------------------------
// FastNGP-NeRF fused forward for gfx1250 (MI455X), wave32 + WMMA f16.
// One wave processes a 16-point tile end-to-end:
//   hash-grid encode -> [WMMA] density MLP -> dir encode -> [WMMA] color MLP.
// Hash tables (64MB) are L2-resident (192MB L2); gathers dominate, MLPs go to
// v_wmma_f32_16x16x32_f16. Intra-wave LDS handoff uses s_wait_dscnt (CDNA5).
// ---------------------------------------------------------------------------

typedef __attribute__((ext_vector_type(16))) _Float16 v16h;
typedef __attribute__((ext_vector_type(8)))  _Float16 v8h;
typedef __attribute__((ext_vector_type(8)))  float    v8f;

#define WAVES       4
#define TILE        16
#define TABLE_SIZE  (1 << 19)
#define TABLE_MASK  (TABLE_SIZE - 1)

// CDNA5 split-counter wait: make this wave's LDS stores visible to all lanes
// of the same wave (wave executes in lockstep; no block barrier needed).
__device__ __forceinline__ void wave_lds_fence() {
  asm volatile("s_wait_dscnt 0" ::: "memory");
}

// Build the f16 16x32 A-operand for one lane (contiguous K, matching B):
// lanes 0-15 hold K = {kb..kb+15} with kb = chunk base,
// lanes 16-31 get kb += 16 (caller passes kb = kchunk + (lane>>4)*16).
__device__ __forceinline__ v16h load_A(const _Float16* row, int kb) {
  v8h lo = *(const v8h*)(row + kb);
  v8h hp = *(const v8h*)(row + kb + 8);
  v16h a;
#pragma unroll
  for (int i = 0; i < 8; ++i) { a[i] = lo[i]; a[8 + i] = hp[i]; }
  return a;
}

__device__ __forceinline__ float softplusf(float x) {
  return fmaxf(x, 0.f) + log1pf(expf(-fabsf(x)));
}
__device__ __forceinline__ float sigmoidf(float x) {
  return 1.f / (1.f + expf(-x));
}

__global__ __launch_bounds__(WAVES * 32)
void ngp_nerf_fused(const float* __restrict__ X,   const float* __restrict__ VD,
                    const float* __restrict__ TAB,
                    const float* __restrict__ Wd1, const float* __restrict__ bd1,
                    const float* __restrict__ Wd2, const float* __restrict__ bd2,
                    const float* __restrict__ Wc1, const float* __restrict__ bc1,
                    const float* __restrict__ Wc2, const float* __restrict__ bc2,
                    const float* __restrict__ Wskip,
                    float* __restrict__ out_rgb,   float* __restrict__ out_sigma)
{
  // Weights (f16) shared by block; per-wave activation tiles.
  __shared__ __align__(32) _Float16 sWd1[64 * 32];   // [hidden][k=32]
  __shared__ __align__(32) _Float16 sWd2[16 * 64];   // [out16][k=64]
  __shared__ __align__(32) _Float16 sWc1[64 * 64];   // Wc1+[Wskip,0], K padded 42->64
  __shared__ __align__(32) _Float16 sWc2[16 * 64];   // Wc2 padded N 3->16
  __shared__ float sBd1[64], sBd2[16], sBc1[64], sBc2[4];
  __shared__ __align__(32) _Float16 sXe[WAVES][16 * 32];  // hash features (A of GEMM1)
  __shared__ __align__(32) _Float16 sH [WAVES][16 * 64];  // density hidden
  __shared__ __align__(32) _Float16 sCI[WAVES][16 * 64];  // color_in (feat+direnc+pad)
  __shared__ __align__(32) _Float16 sHC[WAVES][16 * 64];  // color hidden

  const int tid = threadIdx.x;

  // -------- stage weights: f32 -> f16, fold Wskip into Wc1, pad --------
  for (int i = tid; i < 64 * 32; i += blockDim.x) sWd1[i] = (_Float16)Wd1[i];
  for (int i = tid; i < 16 * 64; i += blockDim.x) sWd2[i] = (_Float16)Wd2[i];
  for (int i = tid; i < 64 * 64; i += blockDim.x) {
    int h = i >> 6, k = i & 63;
    float v = (k < 42) ? Wc1[h * 42 + k] : 0.f;
    if (k < 15) v += Wskip[h * 15 + k];
    sWc1[i] = (_Float16)v;
  }
  for (int i = tid; i < 16 * 64; i += blockDim.x) {
    int n = i >> 6, k = i & 63;
    sWc2[i] = (_Float16)((n < 3) ? Wc2[n * 64 + k] : 0.f);
  }
  for (int i = tid; i < 64; i += blockDim.x) { sBd1[i] = bd1[i]; sBc1[i] = bc1[i]; }
  if (tid < 16) sBd2[tid] = bd2[tid];
  if (tid < 4)  sBc2[tid] = (tid < 3) ? bc2[tid] : 0.f;
  __syncthreads();

  const int wave = tid >> 5;
  const int lane = tid & 31;
  const int m    = lane & 15;   // point row within tile / N-col within 16-tile
  const int hw   = lane >> 4;   // half-wave select

  const long long tp = ((long long)blockIdx.x * WAVES + wave) * TILE;

  _Float16* xe = sXe[wave];
  _Float16* Hs = sH [wave];
  _Float16* CI = sCI[wave];
  _Float16* HC = sHC[wave];

  // ================= hash-grid encode =================
  // lane -> point m; half-wave -> levels 0-7 or 8-15. 8 corners per level.
  {
    const float px = X[(tp + m) * 3 + 0];
    const float py = X[(tp + m) * 3 + 1];
    const float pz = X[(tp + m) * 3 + 2];
#pragma unroll
    for (int li = 0; li < 8; ++li) {
      const int resLo[8] = {16, 20, 25, 32, 40, 50, 64, 80};
      const int resHi[8] = {101, 128, 161, 203, 256, 322, 406, 512};
      const int   l   = hw * 8 + li;
      const float res = (float)(hw ? resHi[li] : resLo[li]);
      float fx = px * res, fy = py * res, fz = pz * res;
      float x0 = floorf(fx), y0 = floorf(fy), z0 = floorf(fz);
      float tx = fx - x0, ty = fy - y0, tz = fz - z0;
      unsigned cx = (unsigned)(int)x0, cy = (unsigned)(int)y0, cz = (unsigned)(int)z0;
      const float2* tab = (const float2*)TAB + (size_t)l * TABLE_SIZE;
      float f0 = 0.f, f1 = 0.f;
#pragma unroll
      for (int c = 0; c < 8; ++c) {
        const unsigned ox = (c >> 2) & 1, oy = (c >> 1) & 1, oz = c & 1;
        unsigned h = (cx + ox) ^ (cy + oy) * 2654435761u ^ (cz + oz) * 805459861u;
        float2 f = tab[h & TABLE_MASK];
        float  w = (ox ? tx : 1.f - tx) * (oy ? ty : 1.f - ty) * (oz ? tz : 1.f - tz);
        f0 = fmaf(w, f.x, f0);
        f1 = fmaf(w, f.y, f1);
      }
      xe[m * 32 + 2 * l]     = (_Float16)f0;
      xe[m * 32 + 2 * l + 1] = (_Float16)f1;
    }
  }
  wave_lds_fence();

  // ================= GEMM1: H = relu(Xe(16x32) @ Wd1^T + bd1) =================
  {
    v16h a = load_A(xe + m * 32, hw * 16);
#pragma unroll
    for (int j = 0; j < 4; ++j) {
      v16h b = *(const v16h*)(sWd1 + (16 * j + m) * 32 + hw * 16);
      v8f  c = {};
      c = __builtin_amdgcn_wmma_f32_16x16x32_f16(false, a, false, b, (short)0, c, false, false);
      const float bias = sBd1[16 * j + m];
#pragma unroll
      for (int r = 0; r < 8; ++r)
        Hs[(hw * 8 + r) * 64 + 16 * j + m] = (_Float16)fmaxf(c[r] + bias, 0.f);
    }
  }
  wave_lds_fence();

  // ================= GEMM2: raw(16x16) = H(16x64) @ Wd2^T + bd2 =================
  {
    v8f c = {};
#pragma unroll
    for (int kc = 0; kc < 2; ++kc) {
      v16h a = load_A(Hs + m * 64, kc * 32 + hw * 16);
      v16h b = *(const v16h*)(sWd2 + m * 64 + kc * 32 + hw * 16);
      c = __builtin_amdgcn_wmma_f32_16x16x32_f16(false, a, false, b, (short)0, c, false, false);
    }
    const float bias = sBd2[m];
    if (m == 0) {           // sigma = softplus(raw[:,0] + 1.5)
#pragma unroll
      for (int r = 0; r < 8; ++r)
        out_sigma[tp + hw * 8 + r] = softplusf(c[r] + bias + 1.5f);
    } else {                // features = raw[:,1:16] -> color_in[0..14]
#pragma unroll
      for (int r = 0; r < 8; ++r)
        CI[(hw * 8 + r) * 64 + (m - 1)] = (_Float16)(c[r] + bias);
    }
  }

  // ============ direction encoding -> CI[.,15..41], zero pad 42..63 ============
  if (hw == 0) {
    float vx = VD[(tp + m) * 3 + 0];
    float vy = VD[(tp + m) * 3 + 1];
    float vz = VD[(tp + m) * 3 + 2];
    const float inv = rsqrtf(vx * vx + vy * vy + vz * vz);
    vx *= inv; vy *= inv; vz *= inv;
    _Float16* row = CI + m * 64;
    row[15] = (_Float16)vx; row[16] = (_Float16)vy; row[17] = (_Float16)vz;
    float fr = 1.f;
#pragma unroll
    for (int f = 0; f < 4; ++f) {
      row[18 + 3 * f + 0] = (_Float16)sinf(vx * fr);
      row[18 + 3 * f + 1] = (_Float16)sinf(vy * fr);
      row[18 + 3 * f + 2] = (_Float16)sinf(vz * fr);
      row[30 + 3 * f + 0] = (_Float16)cosf(vx * fr);
      row[30 + 3 * f + 1] = (_Float16)cosf(vy * fr);
      row[30 + 3 * f + 2] = (_Float16)cosf(vz * fr);
      fr *= 2.f;
    }
  } else {
    _Float16* row = CI + m * 64;
#pragma unroll
    for (int k = 42; k < 64; ++k) row[k] = (_Float16)0.f;
  }
  wave_lds_fence();

  // ========= GEMM3: HC = relu(CI(16x64) @ (Wc1+[Wskip,0])^T + bc1) =========
#pragma unroll
  for (int j = 0; j < 4; ++j) {
    v8f c = {};
#pragma unroll
    for (int kc = 0; kc < 2; ++kc) {
      v16h a = load_A(CI + m * 64, kc * 32 + hw * 16);
      v16h b = *(const v16h*)(sWc1 + (16 * j + m) * 64 + kc * 32 + hw * 16);
      c = __builtin_amdgcn_wmma_f32_16x16x32_f16(false, a, false, b, (short)0, c, false, false);
    }
    const float bias = sBc1[16 * j + m];
#pragma unroll
    for (int r = 0; r < 8; ++r)
      HC[(hw * 8 + r) * 64 + 16 * j + m] = (_Float16)fmaxf(c[r] + bias, 0.f);
  }
  wave_lds_fence();

  // ========= GEMM4: rgb = sigmoid(HC(16x64) @ Wc2p^T + bc2), cols 0..2 =========
  {
    v8f c = {};
#pragma unroll
    for (int kc = 0; kc < 2; ++kc) {
      v16h a = load_A(HC + m * 64, kc * 32 + hw * 16);
      v16h b = *(const v16h*)(sWc2 + m * 64 + kc * 32 + hw * 16);
      c = __builtin_amdgcn_wmma_f32_16x16x32_f16(false, a, false, b, (short)0, c, false, false);
    }
    if (m < 3) {
      const float bias = sBc2[m];
#pragma unroll
      for (int r = 0; r < 8; ++r)
        out_rgb[(tp + hw * 8 + r) * 3 + m] = sigmoidf(c[r] + bias);
    }
  }
}

extern "C" void kernel_launch(void* const* d_in, const int* in_sizes, int n_in,
                              void* d_out, int out_size, void* d_ws, size_t ws_size,
                              hipStream_t stream) {
  (void)n_in; (void)d_ws; (void)ws_size; (void)out_size;
  const float* x     = (const float*)d_in[0];
  const float* vd    = (const float*)d_in[1];
  const float* tabs  = (const float*)d_in[2];
  const float* Wd1   = (const float*)d_in[3];
  const float* bd1   = (const float*)d_in[4];
  const float* Wd2   = (const float*)d_in[5];
  const float* bd2   = (const float*)d_in[6];
  const float* Wc1   = (const float*)d_in[7];
  const float* bc1   = (const float*)d_in[8];
  const float* Wc2   = (const float*)d_in[9];
  const float* bc2   = (const float*)d_in[10];
  const float* Wskip = (const float*)d_in[11];

  const int n = in_sizes[0] / 3;                 // 2,097,152 points
  float* out_rgb   = (float*)d_out;              // [n*3]
  float* out_sigma = (float*)d_out + (size_t)3 * n;  // [n]

  const int pts_per_block = TILE * WAVES;        // 64
  const int blocks = n / pts_per_block;          // 32768 (n is a multiple)

  ngp_nerf_fused<<<blocks, WAVES * 32, 0, stream>>>(
      x, vd, tabs, Wd1, bd1, Wd2, bd2, Wc1, bc1, Wc2, bc2, Wskip,
      out_rgb, out_sigma);
}
